// JKExpert_2310692405504
// MI455X (gfx1250) — matrix operand order repaired and checked by reading, not compiled, not run on hardware
//
#include <hip/hip_runtime.h>
#include <hip/hip_bf16.h>

typedef __bf16 bf16_t;
typedef __attribute__((ext_vector_type(16))) __bf16 v16bf;
typedef __attribute__((ext_vector_type(8)))  __bf16 v8bf;
typedef __attribute__((ext_vector_type(8)))  float  v8f;

#define GCN_N 10000
#define GCN_E 320000
#define GCN_D 256
#define GCN_L 3
#define GCN_K3 768
#define GCN_EPS 1e-5f

// ---------------- degree / norm precompute ----------------

__global__ void __launch_bounds__(256) deg_init_kernel(float* deg, int n) {
    int i = blockIdx.x * blockDim.x + threadIdx.x;
    if (i < n) deg[i] = 1.0f;  // self loop
}

__global__ void __launch_bounds__(256) deg_count_kernel(const int* __restrict__ dst,
                                                        float* __restrict__ deg, int e) {
    for (int i = blockIdx.x * blockDim.x + threadIdx.x; i < e;
         i += gridDim.x * blockDim.x) {
        unsafeAtomicAdd(&deg[dst[i]], 1.0f);
    }
}

__global__ void __launch_bounds__(256) dinv_kernel(float* dinv, int n) {
    int i = blockIdx.x * blockDim.x + threadIdx.x;
    if (i < n) {
        float d = dinv[i];
        dinv[i] = (d > 0.0f) ? __frsqrt_rn(d) : 0.0f;
    }
}

__global__ void __launch_bounds__(256) edge_norm_kernel(const int* __restrict__ src,
                                                        const int* __restrict__ dst,
                                                        const float* __restrict__ dinv,
                                                        float* __restrict__ norm, int e) {
    for (int i = blockIdx.x * blockDim.x + threadIdx.x; i < e;
         i += gridDim.x * blockDim.x) {
        norm[i] = dinv[src[i]] * dinv[dst[i]];
    }
}

// ---------------- bf16 conversions ----------------

__global__ void __launch_bounds__(256) cvt_bf16_kernel(const float* __restrict__ in,
                                                       bf16_t* __restrict__ out, int n) {
    int i = blockIdx.x * blockDim.x + threadIdx.x;
    if (i < n) out[i] = (bf16_t)in[i];
}

// W[k*D + n] (row major, KxDout) -> Wt_bf16[n*K + k]
__global__ void __launch_bounds__(256) cvt_wt_bf16_kernel(const float* __restrict__ W,
                                                          bf16_t* __restrict__ Wt,
                                                          int K, int Dout) {
    int total = K * Dout;
    for (int idx = blockIdx.x * blockDim.x + threadIdx.x; idx < total;
         idx += gridDim.x * blockDim.x) {
        int k = idx % K;
        int n = idx / K;
        Wt[(size_t)n * K + k] = (bf16_t)W[(size_t)k * Dout + n];
    }
}

// ---------------- WMMA GEMM: C[M x Dout] = A[M x K] * B[K x Dout] (+bias) ----------------
// A: bf16 row-major with leading dim lda. Bt: bf16, B transposed (Bt[n*K + k]).
// One wave computes one 16x16 output tile via v_wmma_f32_16x16x32_bf16.

__global__ void __launch_bounds__(256)
gemm_bf16_wmma_kernel(const bf16_t* __restrict__ A, int lda,
                      const bf16_t* __restrict__ Bt, int ldb,
                      float* __restrict__ C, int ldc,
                      int M, int K, int nTiles,
                      const float* __restrict__ bias) {
    int wave = threadIdx.x >> 5;
    int tile = blockIdx.x * 8 + wave;
    int totalTiles = (M >> 4) * nTiles;
    if (tile >= totalTiles) return;          // wave-uniform: EXEC stays all-1s for WMMA

    int tm = tile / nTiles;
    int tn = tile - tm * nTiles;
    int lane = threadIdx.x & 31;
    int half = lane >> 4;                    // 0: lanes 0-15, 1: lanes 16-31
    int r16  = lane & 15;

    const bf16_t* Arow = A + (size_t)(tm * 16 + r16) * lda;          // A row = M index
    const bf16_t* Brow = Bt + (size_t)(tn * 16 + r16) * ldb + 16 * half; // Bt row = N index

    v8f c = {};
    for (int kb = 0; kb < K; kb += 32) {
        // A 16x32 bf16 fragment: elem e -> K = kb + 16*half + e (contiguous)
        v16bf a = *(const v16bf*)(Arow + kb + 16 * half);
        // B 32x16 bf16 fragment: elem e -> K = kb + 16*half + e (contiguous, pre-transposed)
        v16bf b = *(const v16bf*)(Brow + kb);
        c = __builtin_amdgcn_wmma_f32_16x16x32_bf16(
                false, a, false, b, (short)0, c, false, false);
    }

    int ncol = tn * 16 + r16;
    float bv = bias ? bias[ncol] : 0.0f;
    float* Crow = C + (size_t)(tm * 16 + 8 * half) * ldc + ncol;
#pragma unroll
    for (int r = 0; r < 8; ++r)              // C/D layout: VGPR r holds rows r and r+8
        Crow[(size_t)r * ldc] = c[r] + bv;
}

// ---------------- aggregation ----------------

// self-loop contribution: agg[i,:] = t[i,:] * dinv[i]^2
__global__ void __launch_bounds__(256) agg_init_kernel(const float* __restrict__ t,
                                                       const float* __restrict__ dinv,
                                                       float* __restrict__ agg, int total) {
    int idx = blockIdx.x * blockDim.x + threadIdx.x;
    if (idx < total) {
        int i = idx >> 8;
        float di = dinv[i];
        agg[idx] = t[idx] * di * di;
    }
}

// scatter: agg[dst,:] += t[src,:] * norm[e]; d is fastest across lanes (coalesced)
__global__ void __launch_bounds__(256) agg_edges_kernel(const int* __restrict__ src,
                                                        const int* __restrict__ dst,
                                                        const float* __restrict__ norm,
                                                        const float* __restrict__ t,
                                                        float* __restrict__ agg, int e) {
    long total = (long)e << 8;
    for (long idx = (long)blockIdx.x * blockDim.x + threadIdx.x; idx < total;
         idx += (long)gridDim.x * blockDim.x) {
        int ed = (int)(idx >> 8);
        int d  = (int)(idx & 255);
        float w = norm[ed];
        float v = t[((long)src[ed] << 8) + d] * w;
        unsafeAtomicAdd(&agg[((long)dst[ed] << 8) + d], v);
    }
}

// bias + BN(eval) + ReLU, write bf16 straight into JK concat buffer [N, 768] at col l*256
__global__ void __launch_bounds__(256)
bn_relu_kernel(const float* __restrict__ agg,
               const float* __restrict__ bs, const float* __restrict__ scale,
               const float* __restrict__ bbias, const float* __restrict__ mean,
               const float* __restrict__ var,
               bf16_t* __restrict__ hcat, int layer, int total) {
    int idx = blockIdx.x * blockDim.x + threadIdx.x;
    if (idx < total) {
        int i = idx >> 8;
        int d = idx & 255;
        float v = agg[idx] + bs[d];
        v = (v - mean[d]) * __frsqrt_rn(var[d] + GCN_EPS) * scale[d] + bbias[d];
        v = v > 0.0f ? v : 0.0f;
        hcat[(size_t)i * GCN_K3 + layer * GCN_D + d] = (bf16_t)v;
    }
}

// ---------------- host orchestration ----------------

extern "C" void kernel_launch(void* const* d_in, const int* in_sizes, int n_in,
                              void* d_out, int out_size, void* d_ws, size_t ws_size,
                              hipStream_t stream) {
    (void)in_sizes; (void)n_in; (void)out_size; (void)ws_size;

    const float* x        = (const float*)d_in[0];
    const int*   src      = (const int*)d_in[1];
    const int*   dst      = src + GCN_E;
    const float* Ws       = (const float*)d_in[2];
    const float* bs       = (const float*)d_in[3];
    const float* bn_scale = (const float*)d_in[4];
    const float* bn_bias  = (const float*)d_in[5];
    const float* bn_mean  = (const float*)d_in[6];
    const float* bn_var   = (const float*)d_in[7];
    const float* lin_w    = (const float*)d_in[8];
    const float* lin_b    = (const float*)d_in[9];
    float* out = (float*)d_out;

    // bump-allocate scratch (256B aligned regions)
    size_t off = 0;
    auto alloc = [&](size_t bytes) {
        void* p = (char*)d_ws + off;
        off += (bytes + 255) & ~(size_t)255;
        return p;
    };
    bf16_t* xbf    = (bf16_t*)alloc((size_t)GCN_N * GCN_D * 2);   // x in bf16
    bf16_t* hcat   = (bf16_t*)alloc((size_t)GCN_N * GCN_K3 * 2);  // JK concat, bf16
    bf16_t* Wbf    = (bf16_t*)alloc((size_t)GCN_D * GCN_D * 2);   // W[l]^T bf16 (reused)
    bf16_t* linWbf = (bf16_t*)alloc((size_t)GCN_K3 * GCN_D * 2);  // lin_w^T bf16
    float*  t      = (float*)alloc((size_t)GCN_N * GCN_D * 4);    // h @ W
    float*  agg    = (float*)alloc((size_t)GCN_N * GCN_D * 4);    // aggregated
    float*  dinv   = (float*)alloc((size_t)GCN_N * 4);            // deg -> deg^-1/2
    float*  norm   = (float*)alloc((size_t)GCN_E * 4);            // per-edge norm

    const int TB = 256;
    const int ND = GCN_N * GCN_D;

    // degrees + per-edge norms (once)
    deg_init_kernel<<<(GCN_N + TB - 1) / TB, TB, 0, stream>>>(dinv, GCN_N);
    deg_count_kernel<<<1024, TB, 0, stream>>>(dst, dinv, GCN_E);
    dinv_kernel<<<(GCN_N + TB - 1) / TB, TB, 0, stream>>>(dinv, GCN_N);
    edge_norm_kernel<<<1024, TB, 0, stream>>>(src, dst, dinv, norm, GCN_E);

    // x -> bf16
    cvt_bf16_kernel<<<(ND + TB - 1) / TB, TB, 0, stream>>>(x, xbf, ND);

    const int tiles256 = (GCN_N / 16) * (GCN_D / 16);   // 10000 tiles
    const int gemmBlocks = (tiles256 + 7) / 8;

    for (int l = 0; l < GCN_L; ++l) {
        // W[l] -> transposed bf16
        cvt_wt_bf16_kernel<<<256, TB, 0, stream>>>(Ws + (size_t)l * GCN_D * GCN_D,
                                                   Wbf, GCN_D, GCN_D);
        // t = h @ W[l]
        const bf16_t* Ain = (l == 0) ? xbf : (hcat + (size_t)(l - 1) * GCN_D);
        int lda = (l == 0) ? GCN_D : GCN_K3;
        gemm_bf16_wmma_kernel<<<gemmBlocks, TB, 0, stream>>>(
            Ain, lda, Wbf, GCN_D, t, GCN_D, GCN_N, GCN_D, GCN_D / 16, nullptr);
        // aggregate: self loops then edges
        agg_init_kernel<<<(ND + TB - 1) / TB, TB, 0, stream>>>(t, dinv, agg, ND);
        agg_edges_kernel<<<16384, TB, 0, stream>>>(src, dst, norm, t, agg, GCN_E);
        // bias + BN + ReLU -> bf16 into concat buffer
        bn_relu_kernel<<<(ND + TB - 1) / TB, TB, 0, stream>>>(
            agg, bs + (size_t)l * GCN_D, bn_scale + (size_t)l * GCN_D,
            bn_bias + (size_t)l * GCN_D, bn_mean + (size_t)l * GCN_D,
            bn_var + (size_t)l * GCN_D, hcat, l, ND);
    }

    // final linear: out = hcat @ lin_w + lin_b
    cvt_wt_bf16_kernel<<<768, TB, 0, stream>>>(lin_w, linWbf, GCN_K3, GCN_D);
    gemm_bf16_wmma_kernel<<<gemmBlocks, TB, 0, stream>>>(
        hcat, GCN_K3, linWbf, GCN_K3, out, GCN_D, GCN_N, GCN_K3, GCN_D / 16, lin_b);
}
